// QatTernaryLinear_541165879839
// MI455X (gfx1250) — compile-verified, matrix operand order checked by reading, not run on hardware
//
#include <hip/hip_runtime.h>
#include <hip/hip_bf16.h>

// ---------------------------------------------------------------------------
// QAT ternary linear:  y = alpha * (x @ Q^T) + bias
//   alpha = mean(|w|), Q = clip(round(w/alpha), -1, 1)  (exact in bf16)
// M=8192 (B*S), K=4096 (DIN), N=16384 (DOUT)
// GEMM: 256x128 block tile, BK=64, TDM double-buffered LDS, bf16 WMMA,
//       8 waves as 4x2 grid of 64x64 wave tiles (16 WMMA per K=32 sub-step).
// ---------------------------------------------------------------------------

#define B_    2
#define S_    4096
#define DIN_  4096
#define DOUT_ 16384
#define M_    (B_ * S_)

#define BM 256
#define BN 128
#define BK 64
#define LDSW 72   // row stride in elements (144B = 128B row + 16B TDM pad)

typedef __attribute__((ext_vector_type(16))) __bf16 v16bf;
typedef __attribute__((ext_vector_type(8)))  __bf16 v8bf;
typedef __attribute__((ext_vector_type(4)))  __bf16 v4bf;
typedef __attribute__((ext_vector_type(8)))  float  v8f;
typedef __attribute__((ext_vector_type(4)))  unsigned int u32x4;
typedef __attribute__((ext_vector_type(8)))  int i32x8;
typedef __attribute__((ext_vector_type(4)))  int i32x4;

// Use the Tensor Data Mover only in the device pass and only if the builtins exist.
#if defined(__HIP_DEVICE_COMPILE__)
#  if __has_builtin(__builtin_amdgcn_tensor_load_to_lds) && \
      __has_builtin(__builtin_amdgcn_s_wait_tensorcnt)
#    define USE_TDM 1
#  endif
#endif
#ifndef USE_TDM
#  define USE_TDM 0
#endif

// ---------------------------------------------------------------------------
// Kernel 1: per-block partial sums of |w| (deterministic, no atomics)
// ---------------------------------------------------------------------------
__global__ __launch_bounds__(256) void k_abs_partial(const float* __restrict__ w,
                                                     float* __restrict__ partials,
                                                     long n4) {
  const float4* w4 = (const float4*)w;
  float s = 0.0f;
  long stride = (long)gridDim.x * blockDim.x;
  for (long i = (long)blockIdx.x * blockDim.x + threadIdx.x; i < n4; i += stride) {
    float4 v = w4[i];
    s += fabsf(v.x) + fabsf(v.y) + fabsf(v.z) + fabsf(v.w);
  }
  __shared__ float red[256];
  red[threadIdx.x] = s;
  __syncthreads();
  for (int off = 128; off > 0; off >>= 1) {
    if ((int)threadIdx.x < off) red[threadIdx.x] += red[threadIdx.x + off];
    __syncthreads();
  }
  if (threadIdx.x == 0) partials[blockIdx.x] = red[0];
}

// ---------------------------------------------------------------------------
// Kernel 2: finalize alpha = sum / count, store [alpha, 1/alpha]
// ---------------------------------------------------------------------------
__global__ __launch_bounds__(256) void k_finalize(const float* __restrict__ partials,
                                                  int nPartials,
                                                  float* __restrict__ scale,
                                                  float invCount) {
  float s = 0.0f;
  for (int i = threadIdx.x; i < nPartials; i += 256) s += partials[i];
  __shared__ float red[256];
  red[threadIdx.x] = s;
  __syncthreads();
  for (int off = 128; off > 0; off >>= 1) {
    if ((int)threadIdx.x < off) red[threadIdx.x] += red[threadIdx.x + off];
    __syncthreads();
  }
  if (threadIdx.x == 0) {
    float alpha = red[0] * invCount;
    scale[0] = alpha;
    scale[1] = 1.0f / alpha;
  }
}

// ---------------------------------------------------------------------------
// Kernel 3: ternary-quantize weight -> bf16 Q in {-1,0,+1}  (exact)
// ---------------------------------------------------------------------------
__global__ __launch_bounds__(256) void k_quant_w(const float* __restrict__ w,
                                                 __bf16* __restrict__ wq,
                                                 const float* __restrict__ scale,
                                                 long n4) {
  const float inva = scale[1];
  const float4* w4 = (const float4*)w;
  v4bf* o4 = (v4bf*)wq;
  long stride = (long)gridDim.x * blockDim.x;
  for (long i = (long)blockIdx.x * blockDim.x + threadIdx.x; i < n4; i += stride) {
    float4 v = w4[i];
    v4bf o;
    o.x = (__bf16)fminf(1.0f, fmaxf(-1.0f, rintf(v.x * inva)));
    o.y = (__bf16)fminf(1.0f, fmaxf(-1.0f, rintf(v.y * inva)));
    o.z = (__bf16)fminf(1.0f, fmaxf(-1.0f, rintf(v.z * inva)));
    o.w = (__bf16)fminf(1.0f, fmaxf(-1.0f, rintf(v.w * inva)));
    o4[i] = o;
  }
}

// ---------------------------------------------------------------------------
// Kernel 4: convert x (f32) -> bf16
// ---------------------------------------------------------------------------
__global__ __launch_bounds__(256) void k_cvt_x(const float* __restrict__ x,
                                               __bf16* __restrict__ x16,
                                               long n4) {
  const float4* x4 = (const float4*)x;
  v4bf* o4 = (v4bf*)x16;
  long stride = (long)gridDim.x * blockDim.x;
  for (long i = (long)blockIdx.x * blockDim.x + threadIdx.x; i < n4; i += stride) {
    float4 v = x4[i];
    v4bf o;
    o.x = (__bf16)v.x; o.y = (__bf16)v.y; o.z = (__bf16)v.z; o.w = (__bf16)v.w;
    o4[i] = o;
  }
}

// ---------------------------------------------------------------------------
// LDS byte address (wave-relative) of a __shared__ object
// ---------------------------------------------------------------------------
__device__ __forceinline__ unsigned lds_addr_of(const void* p) {
  return (unsigned)(unsigned long long)(const __attribute__((address_space(3))) void*)p;
}

#if USE_TDM
// ---------------------------------------------------------------------------
// Issue one TDM tile load: tile_dim0=BK elems (K dir), tile_dim1=tileRows rows,
// 2-byte elements, row stride in tensor = DIN_, HW-padded LDS rows:
// pad_interval = 32 DWORDs (128B = one row), pad_amount = 4 DWORDs (16B)
// => LDS row stride 144B = LDSW elements.
// ---------------------------------------------------------------------------
__device__ __forceinline__ void tdm_load_tile(const void* gsrc, unsigned ldsByte,
                                              unsigned rowsTotal, unsigned tileRows) {
  unsigned long long ga = (unsigned long long)gsrc;
  u32x4 g0;
  g0.x = 1u;                                     // count=1 (valid), user mode
  g0.y = ldsByte;                                // lds_addr
  g0.z = (unsigned)ga;                           // global_addr[31:0]
  g0.w = (unsigned)((ga >> 32) & 0x01FFFFFFu)    // global_addr[56:32]
       | 0x80000000u;                            // type = 2 ("image")
  i32x8 g1;
  g1[0] = (int)((1u << 16)      // data_size = 1 -> 2 bytes
              | (1u << 20)      // pad_enable
              | (4u << 22)      // pad_interval: 32 DWORDs
              | (3u << 25));    // pad_amount:   4 DWORDs
  g1[1] = (int)(((unsigned)DIN_ & 0xFFFFu) << 16);          // tensor_dim0[15:0]
  g1[2] = (int)((((unsigned)DIN_ >> 16) & 0xFFFFu)          // tensor_dim0[31:16]
              | ((rowsTotal & 0xFFFFu) << 16));             // tensor_dim1[15:0]
  g1[3] = (int)(((rowsTotal >> 16) & 0xFFFFu)               // tensor_dim1[31:16]
              | ((unsigned)BK << 16));                      // tile_dim0 = 64
  g1[4] = (int)(tileRows & 0xFFFFu);                        // tile_dim1, tile_dim2=0
  g1[5] = (int)DIN_;                                        // tensor_dim0_stride[31:0]
  g1[6] = 0;                                                // stride hi / dim1_stride lo
  g1[7] = 0;
  i32x4 gz = {0, 0, 0, 0};
#if __clang_major__ >= 23
  i32x8 gz8 = {0, 0, 0, 0, 0, 0, 0, 0};
  __builtin_amdgcn_tensor_load_to_lds(g0, g1, gz, gz, gz8, 0);
#else
  __builtin_amdgcn_tensor_load_to_lds(g0, g1, gz, gz, 0);
#endif
}
#endif // USE_TDM

// ---------------------------------------------------------------------------
// One LDS stage of WMMAs: wave tile 64x64, BK=64 -> 2 x (16 WMMA) = 32 WMMA
// ---------------------------------------------------------------------------
__device__ __forceinline__ void mma_stage(const __bf16* tA, const __bf16* tB,
                                          int wm, int wn, int rSel, int kSel,
                                          v8f (&acc)[4][4]) {
#pragma unroll
  for (int ks = 0; ks < BK; ks += 32) {
    v16bf aF[4];
    v16bf bF[4];
#pragma unroll
    for (int i = 0; i < 4; i++) {
      const __bf16* p = tA + (wm + i * 16 + rSel) * LDSW + ks + kSel;
      union { v16bf v; v8bf h[2]; } u;
      u.h[0] = *(const v8bf*)p;
      u.h[1] = *(const v8bf*)(p + 16);
      aF[i] = u.v;
    }
#pragma unroll
    for (int j = 0; j < 4; j++) {
      const __bf16* p = tB + (wn + j * 16 + rSel) * LDSW + ks + kSel;
      union { v16bf v; v8bf h[2]; } u;
      u.h[0] = *(const v8bf*)p;
      u.h[1] = *(const v8bf*)(p + 16);
      bF[j] = u.v;
    }
#pragma unroll
    for (int i = 0; i < 4; i++)
#pragma unroll
      for (int j = 0; j < 4; j++)
        acc[i][j] = __builtin_amdgcn_wmma_f32_16x16x32_bf16(
            false, aF[i], false, bF[j], (short)0, acc[i][j], false, false);
  }
}

// ---------------------------------------------------------------------------
// Kernel 5: tiled bf16 WMMA GEMM  C[M,N] = alpha*(A[M,K] @ B[N,K]^T) + bias[N]
// ---------------------------------------------------------------------------
__global__ __launch_bounds__(256) void k_gemm(const __bf16* __restrict__ A,
                                              const __bf16* __restrict__ Bw,
                                              const float* __restrict__ bias,
                                              const float* __restrict__ scale,
                                              float* __restrict__ C) {
  constexpr int K = DIN_;
  constexpr int N = DOUT_;

  __shared__ __align__(128) __bf16 sA[2][BM * LDSW];   // 2 x 36864 B
  __shared__ __align__(128) __bf16 sB[2][BN * LDSW];   // 2 x 18432 B

  const int tid  = threadIdx.x;
  const int wave = tid >> 5;
  const int lane = tid & 31;

  const int bn0 = blockIdx.x * BN;
  const int bm0 = blockIdx.y * BM;

  const int wm = (wave >> 1) * 64;   // wave M offset: 0/64/128/192
  const int wn = (wave & 1) * 64;    // wave N offset: 0/64

  v8f acc[4][4];
#pragma unroll
  for (int i = 0; i < 4; i++)
#pragma unroll
    for (int j = 0; j < 4; j++) acc[i][j] = (v8f)(0.0f);

  const float alpha = scale[0];

  // 16-bit A/B fragment layout: lanes 0-15 hold K {0..7,16..23},
  // lanes 16-31 hold K {8..15,24..31} (per 32-wide K sub-step)
  const int kSel = (lane < 16) ? 0 : 8;
  const int rSel = lane & 15;

#if USE_TDM
  const unsigned aBase = lds_addr_of(&sA[0][0]);
  const unsigned bBase = lds_addr_of(&sB[0][0]);
  const unsigned aBufBytes = (unsigned)(BM * LDSW * 2);
  const unsigned bBufBytes = (unsigned)(BN * LDSW * 2);

  // Uniform (SGPR) test so the TDM-issuer branch is an s_cbranch, not exec mask.
  const bool isWave0 = (__builtin_amdgcn_readfirstlane(tid) < 32);

  if (isWave0) {
    tdm_load_tile(A  + (size_t)bm0 * K, aBase, (unsigned)M_,    (unsigned)BM);
    tdm_load_tile(Bw + (size_t)bn0 * K, bBase, (unsigned)DOUT_, (unsigned)BN);
  }

  int buf = 0;
  for (int k0 = 0; k0 < K; k0 += BK) {
    if (isWave0) {
      if (k0 + BK < K) {
        // issue next tile into the other buffer, then wait for current tile
        tdm_load_tile(A  + (size_t)bm0 * K + (k0 + BK),
                      aBase + (unsigned)(buf ^ 1) * aBufBytes,
                      (unsigned)M_, (unsigned)BM);
        tdm_load_tile(Bw + (size_t)bn0 * K + (k0 + BK),
                      bBase + (unsigned)(buf ^ 1) * bBufBytes,
                      (unsigned)DOUT_, (unsigned)BN);
        __builtin_amdgcn_s_wait_tensorcnt(2);
      } else {
        __builtin_amdgcn_s_wait_tensorcnt(0);
      }
    }
    __syncthreads();                       // current buffer ready for all waves
    mma_stage(&sA[0][0] + buf * (BM * LDSW), &sB[0][0] + buf * (BN * LDSW),
              wm, wn, rSel, kSel, acc);
    __syncthreads();                       // reads done before TDM overwrites
    buf ^= 1;
  }
#else
  // Fallback: synchronous global->VGPR->LDS staging (single buffer)
  const int aRow = tid;                 // A: one thread per row, 64 elems
  const int bRow = tid >> 1;            // B: two threads per row
  const int bK   = (tid & 1) * 32;

  const __bf16* gA = A  + (size_t)(bm0 + aRow) * K;
  const __bf16* gB = Bw + (size_t)(bn0 + bRow) * K + bK;

  for (int k0 = 0; k0 < K; k0 += BK) {
    if (k0 + BK < K) {
      __builtin_prefetch(gA + k0 + BK, 0, 1);
      __builtin_prefetch(gB + k0 + BK, 0, 1);
    }
    v8bf a[8], b[4];
#pragma unroll
    for (int c = 0; c < 8; c++) a[c] = *(const v8bf*)(gA + k0 + c * 8);
#pragma unroll
    for (int c = 0; c < 4; c++) b[c] = *(const v8bf*)(gB + k0 + c * 8);

    __syncthreads();
    __bf16* dA = &sA[0][0] + aRow * LDSW;
    __bf16* dB = &sB[0][0] + bRow * LDSW + bK;
#pragma unroll
    for (int c = 0; c < 8; c++) *(v8bf*)(dA + c * 8) = a[c];
#pragma unroll
    for (int c = 0; c < 4; c++) *(v8bf*)(dB + c * 8) = b[c];
    __syncthreads();

    mma_stage(&sA[0][0], &sB[0][0], wm, wn, rSel, kSel, acc);
  }
#endif

  // Epilogue: C/D layout — lanes 0-15: M=r, N=lane; lanes 16-31: M=r+8, N=lane-16
  const int colBase = bn0 + wn + (lane & 15);
  const int rowOff  = (lane >> 4) * 8;
#pragma unroll
  for (int j = 0; j < 4; j++) {
    const float bj = bias[colBase + j * 16];
#pragma unroll
    for (int i = 0; i < 4; i++) {
      const int rowG = bm0 + wm + i * 16 + rowOff;
#pragma unroll
      for (int r = 0; r < 8; r++) {
        C[(size_t)(rowG + r) * N + colBase + j * 16] = alpha * acc[i][j][r] + bj;
      }
    }
  }
}

// ---------------------------------------------------------------------------
// Launch
// ---------------------------------------------------------------------------
extern "C" void kernel_launch(void* const* d_in, const int* in_sizes, int n_in,
                              void* d_out, int out_size, void* d_ws, size_t ws_size,
                              hipStream_t stream) {
  (void)in_sizes; (void)n_in; (void)out_size; (void)ws_size;

  const float* x    = (const float*)d_in[0];
  const float* w    = (const float*)d_in[1];
  const float* bias = (const float*)d_in[2];
  float* out = (float*)d_out;

  // workspace layout
  char* ws = (char*)d_ws;
  float*  scale    = (float*)ws;                 // [alpha, 1/alpha]
  float*  partials = (float*)(ws + 256);         // 1024 floats
  __bf16* x16      = (__bf16*)(ws + 8192);                               // 64 MB
  __bf16* w16      = (__bf16*)(ws + 8192 + (size_t)M_ * DIN_ * 2);       // 128 MB

  const long nW4 = (long)DOUT_ * DIN_ / 4;
  const long nX4 = (long)M_ * DIN_ / 4;

  k_abs_partial<<<1024, 256, 0, stream>>>(w, partials, nW4);
  k_finalize<<<1, 256, 0, stream>>>(partials, 1024, scale,
                                    1.0f / (float)((size_t)DOUT_ * DIN_));
  k_quant_w<<<4096, 256, 0, stream>>>(w, w16, scale, nW4);
  k_cvt_x<<<2048, 256, 0, stream>>>(x, x16, nX4);

  dim3 grid(DOUT_ / BN, M_ / BM);   // 128 x 32
  k_gemm<<<grid, 256, 0, stream>>>(x16, w16, bias, scale, out);
}
